// OptionPolicyV2_80135499808822
// MI455X (gfx1250) — compile-verified, matrix operand order checked by reading, not run on hardware
//
#include <hip/hip_runtime.h>
#include <math.h>

// ---------------------------------------------------------------------------
// Types for CDNA5 WMMA (bf16 A/B, f32 accum)
// ---------------------------------------------------------------------------
typedef __attribute__((ext_vector_type(16))) __bf16 v16bf;
typedef __attribute__((ext_vector_type(8)))  __bf16 bf16x8;
typedef __attribute__((ext_vector_type(4)))  __bf16 bf16x4;
typedef __attribute__((ext_vector_type(8)))  float  v8f;

#define LC 512
#define NEG_INF (-1e30f)

// packed-weight element offsets inside workspace (bf16 elements)
#define W1P 0
#define W2P 16384
#define W3P 81920
#define W1O 114688
#define W2O 131072
#define W3O 196608
#define WPACK_BYTES 532480   // 266240 bf16 elements

// ---------------------------------------------------------------------------
// Device helpers
// ---------------------------------------------------------------------------
__device__ __forceinline__ v8f splat8(float x) {
  v8f v;
  #pragma unroll
  for (int i = 0; i < 8; ++i) v[i] = x;
  return v;
}

__device__ __forceinline__ v8f wmma_bf16(v16bf a, v16bf b, v8f c) {
  return __builtin_amdgcn_wmma_f32_16x16x32_bf16(
      /*neg_a=*/false, a, /*neg_b=*/false, b,
      /*c_mod=*/(short)0, c, /*reuse_a=*/false, /*reuse_b=*/false);
}

// B fragment: packed layout, 32 bytes per lane, contiguous
__device__ __forceinline__ v16bf load_bfrag(const __bf16* __restrict__ base, int tile, int lane) {
  const __bf16* p = base + ((size_t)tile << 9) + (lane << 4);
  union { v16bf v; bf16x8 h[2]; } u;
  u.h[0] = *(const bf16x8*)p;
  u.h[1] = *(const bf16x8*)(p + 8);
  return u.v;
}

// A fragment from LDS (row-major [16][K], stride in elements):
// lane row = lane&15; element e -> k = kt*32 + 8*(lane>>4) + (e<8 ? e : e+8)
__device__ __forceinline__ v16bf load_afrag(const __bf16* base, int stride, int kt, int lane) {
  const int n = lane & 15, half = lane >> 4;
  const __bf16* p = base + n * stride + kt * 32 + half * 8;
  union { v16bf v; bf16x8 h[2]; } u;
  u.h[0] = *(const bf16x8*)p;        // k = 8*half .. 8*half+7
  u.h[1] = *(const bf16x8*)(p + 16); // k = 16+8*half .. +7
  return u.v;
}

// Two GEMM+ReLU layers: s(16x64) -> h1(16x256) -> h2(16x256) into hA (LDS)
__device__ __forceinline__ void mlp_2layer(const __bf16* __restrict__ w1,
                                           const __bf16* __restrict__ w2,
                                           const float* __restrict__ b1,
                                           const float* __restrict__ b2,
                                           v16bf sa0, v16bf sa1, __bf16* hA, int lane) {
  const int n16 = lane & 15;
  const int half = lane >> 4;
  #pragma unroll 1
  for (int nt = 0; nt < 16; ++nt) {
    v8f acc = splat8(b1[nt * 16 + n16]);
    acc = wmma_bf16(sa0, load_bfrag(w1, nt * 2 + 0, lane), acc);
    acc = wmma_bf16(sa1, load_bfrag(w1, nt * 2 + 1, lane), acc);
    #pragma unroll
    for (int r = 0; r < 8; ++r)
      hA[(r + 8 * half) * 264 + nt * 16 + n16] = (__bf16)fmaxf(acc[r], 0.f);
  }
  v8f acc2[16];
  #pragma unroll
  for (int nt = 0; nt < 16; ++nt) acc2[nt] = splat8(b2[nt * 16 + n16]);
  #pragma unroll 1
  for (int kt = 0; kt < 8; ++kt) {
    v16bf af = load_afrag(hA, 264, kt, lane);
    #pragma unroll
    for (int nt = 0; nt < 16; ++nt)
      acc2[nt] = wmma_bf16(af, load_bfrag(w2, nt * 8 + kt, lane), acc2[nt]);
  }
  #pragma unroll
  for (int nt = 0; nt < 16; ++nt) {
    #pragma unroll
    for (int r = 0; r < 8; ++r)
      hA[(r + 8 * half) * 264 + nt * 16 + n16] = (__bf16)fmaxf(acc2[nt][r], 0.f);
  }
}

// ---------------------------------------------------------------------------
// Weight packing: W (fin x fout, row-major f32) -> WMMA B fragments (bf16)
// ---------------------------------------------------------------------------
__global__ void k_pack(const float* __restrict__ W, __bf16* __restrict__ dst, int fin, int fout) {
  int idx = blockIdx.x * 256 + threadIdx.x;
  int KT = fin >> 5;
  int total = (fout >> 4) * KT * 512;
  if (idx >= total) return;
  int e    = idx & 15;
  int lane = (idx >> 4) & 31;
  int tile = idx >> 9;
  int kt = tile % KT;
  int nt = tile / KT;
  int half = lane >> 4;
  int n = lane & 15;
  int k = kt * 32 + ((e < 8) ? e : (e + 8)) + 8 * half;
  dst[idx] = (__bf16)W[(size_t)k * fout + nt * 16 + n];
}

// ---------------------------------------------------------------------------
// Fused MLP kernel: 4 waves/block, one 16-timestep tile per wave.
// Writes log_trs, log_pis, entropy, and log_tr0 (workspace).
// ---------------------------------------------------------------------------
__global__ __launch_bounds__(128) void k_mlp(
    const float* __restrict__ s_arr, const float* __restrict__ a_arr,
    const float* __restrict__ pb1, const float* __restrict__ pb2, const float* __restrict__ pb3,
    const float* __restrict__ ob1, const float* __restrict__ ob2, const float* __restrict__ ob3,
    const float* __restrict__ a_log_std, const __bf16* __restrict__ wpack,
    float* __restrict__ out_ltrs, float* __restrict__ out_lpis, float* __restrict__ out_ent,
    float* __restrict__ ws_logtr0, int T)
{
  __shared__ __attribute__((aligned(16))) __bf16 sbuf[64 * 72];   // 64 rows x 64 cols (pad 72)
  __shared__ __attribute__((aligned(16))) __bf16 hbuf[64 * 264];  // 64 rows x 256 cols (pad 264)
  __shared__ float abuf[64 * 8];

  const int tid  = threadIdx.x;
  const int lane = tid & 31;
  const int wv   = tid >> 5;
  const int half = lane >> 4;
  const int n16  = lane & 15;
  const int row0 = blockIdx.x * 64;
  const int wrow = wv * 16;
  const int tbase = row0 + wrow;

  // warm L2 with the packed-weight stream (global_prefetch_b8)
  #pragma unroll 1
  for (int i = tid; i < 1024; i += 128)
    __builtin_prefetch((const void*)(wpack + (size_t)i * 256), 0, 1);

  // stage s tile as bf16 and a tile as f32
  const float4* s4 = (const float4*)s_arr + (size_t)row0 * 16;
  #pragma unroll
  for (int i = 0; i < 8; ++i) {
    int idx = tid + i * 128;
    float4 v = s4[idx];
    int r = idx >> 4, c4 = idx & 15;
    bf16x4 o;
    o[0] = (__bf16)v.x; o[1] = (__bf16)v.y; o[2] = (__bf16)v.z; o[3] = (__bf16)v.w;
    *(bf16x4*)(sbuf + r * 72 + c4 * 4) = o;
  }
  #pragma unroll
  for (int i = 0; i < 4; ++i) {
    int idx = tid + i * 128;
    abuf[idx] = a_arr[(size_t)row0 * 8 + idx];
  }
  __syncthreads();

  __bf16* hA = hbuf + wrow * 264;
  const __bf16* sA = sbuf + wrow * 72;

  v16bf sa0 = load_afrag(sA, 72, 0, lane);
  v16bf sa1 = load_afrag(sA, 72, 1, lane);

  float at8[8];
  #pragma unroll
  for (int r = 0; r < 8; ++r)
    at8[r] = abuf[(wrow + r + 8 * half) * 8 + (lane & 7)];
  float als  = a_log_std[lane & 7];
  float lstd = -5.f + 3.5f * (tanhf(als) + 1.f);   // LOG_STD_MIN + 0.5*range*(tanh+1)
  float istd = __expf(-lstd);

  // ===================== policy MLP + Gaussian log-prob =====================
  mlp_2layer(wpack + W1P, wpack + W2P, pb1, pb2, sa0, sa1, hA, lane);

  v16bf aF[8];
  #pragma unroll
  for (int kt = 0; kt < 8; ++kt) aF[kt] = load_afrag(hA, 264, kt, lane);

  #pragma unroll 1
  for (int nt = 0; nt < 8; ++nt) {
    v8f acc = splat8(pb3[nt * 16 + n16]);
    #pragma unroll
    for (int kt = 0; kt < 8; ++kt)
      acc = wmma_bf16(aF[kt], load_bfrag(wpack + W3P, nt * 8 + kt, lane), acc);
    #pragma unroll
    for (int r = 0; r < 8; ++r) {
      float mv = fminf(fmaxf(acc[r], -10.f), 10.f);
      float z = (at8[r] - mv) * istd;
      float e = -0.5f * z * z - lstd - 0.91893853320467274f;  // 0.5*log(2*pi)
      e += __shfl_xor(e, 1);
      e += __shfl_xor(e, 2);
      e += __shfl_xor(e, 4);
      if ((lane & 7) == 0) {
        int t = tbase + r + 8 * half;
        int c = nt * 2 + ((lane >> 3) & 1);
        out_lpis[(size_t)t * 16 + c] = e;
      }
    }
  }

  // ===================== option MLP + log-softmax + entropy =====================
  mlp_2layer(wpack + W1O, wpack + W2O, ob1, ob2, sa0, sa1, hA, lane);
  #pragma unroll
  for (int kt = 0; kt < 8; ++kt) aF[kt] = load_afrag(hA, 264, kt, lane);

  #pragma unroll 1
  for (int nt = 0; nt < 17; ++nt) {
    v8f acc = splat8(ob3[nt * 16 + n16]);
    #pragma unroll
    for (int kt = 0; kt < 8; ++kt)
      acc = wmma_bf16(aF[kt], load_bfrag(wpack + W3O, nt * 8 + kt, lane), acc);
    #pragma unroll
    for (int r = 0; r < 8; ++r) {
      float x = acc[r];
      float mx = x;
      mx = fmaxf(mx, __shfl_xor(mx, 1));
      mx = fmaxf(mx, __shfl_xor(mx, 2));
      mx = fmaxf(mx, __shfl_xor(mx, 4));
      mx = fmaxf(mx, __shfl_xor(mx, 8));
      float ex = __expf(x - mx);
      float sm = ex;
      sm += __shfl_xor(sm, 1);
      sm += __shfl_xor(sm, 2);
      sm += __shfl_xor(sm, 4);
      sm += __shfl_xor(sm, 8);
      float lse = mx + __logf(sm);
      float ls = x - lse;
      int t = tbase + r + 8 * half;
      if (nt < 16) {
        if (t >= 1) out_ltrs[(size_t)(t - 1) * 256 + nt * 16 + n16] = ls;
        float cc = ls * (ex / sm);     // ls * exp(ls)
        cc += __shfl_xor(cc, 1);
        cc += __shfl_xor(cc, 2);
        cc += __shfl_xor(cc, 4);
        cc += __shfl_xor(cc, 8);
        if (n16 == 0 && t >= 1) out_ent[(size_t)(t - 1) * 16 + nt] = -cc;
      } else {
        if (t == 0) ws_logtr0[n16] = ls;   // log_trs_full[0, 16, :]
      }
    }
  }
}

// ---------------------------------------------------------------------------
// Phase 1: per-chunk composite operator in the (logsumexp,+) semiring.
// O_t[i][j] = log_trs[t-1][i][j] + log_pis[t][j];  M_k = O_{t0} o ... o O_{t1}
// ---------------------------------------------------------------------------
__global__ void k_chunk(const float* __restrict__ ltrs, const float* __restrict__ lpis,
                        float* __restrict__ chunkM, int T)
{
  const int k = blockIdx.x;
  const int tid = threadIdx.x;
  const int i = tid >> 4, j = tid & 15;
  __shared__ float M[256];
  __shared__ float O[256];
  M[tid] = (i == j) ? 0.f : NEG_INF;
  int t0 = k * LC + 1;
  int t1 = (k + 1) * LC; if (t1 > T - 1) t1 = T - 1;
  for (int t = t0; t <= t1; ++t) {
    __syncthreads();
    O[tid] = ltrs[(size_t)(t - 1) * 256 + tid] + lpis[(size_t)t * 16 + j];
    __syncthreads();
    float mx = NEG_INF;
    #pragma unroll
    for (int kk = 0; kk < 16; ++kk) mx = fmaxf(mx, M[i * 16 + kk] + O[kk * 16 + j]);
    float s = 0.f;
    #pragma unroll
    for (int kk = 0; kk < 16; ++kk) s += __expf(M[i * 16 + kk] + O[kk * 16 + j] - mx);
    float nv = mx + __logf(s);
    __syncthreads();
    M[tid] = nv;
  }
  __syncthreads();
  chunkM[(size_t)k * 256 + tid] = M[tid];
}

// Phase 2 forward: boundary alphas across chunks
__global__ void k_scan_f(const float* __restrict__ chunkM, const float* __restrict__ logtr0,
                         const float* __restrict__ lpis, float* __restrict__ Avec, int C)
{
  const int j = threadIdx.x;
  __shared__ float av[16];
  if (j < 16) {
    float a0 = logtr0[j] + lpis[j];
    av[j] = a0;
    Avec[j] = a0;
  }
  __syncthreads();
  for (int k = 0; k < C; ++k) {
    float nv = 0.f;
    if (j < 16) {
      const float* M = chunkM + (size_t)k * 256;
      float mx = NEG_INF;
      #pragma unroll
      for (int i = 0; i < 16; ++i) mx = fmaxf(mx, av[i] + M[i * 16 + j]);
      float s = 0.f;
      #pragma unroll
      for (int i = 0; i < 16; ++i) s += __expf(av[i] + M[i * 16 + j] - mx);
      nv = mx + __logf(s);
    }
    __syncthreads();
    if (j < 16) { av[j] = nv; Avec[(size_t)(k + 1) * 16 + j] = nv; }
    __syncthreads();
  }
}

// Phase 2 backward: boundary betas across chunks (same matrices, other side)
__global__ void k_scan_b(const float* __restrict__ chunkM, float* __restrict__ Bvec, int C)
{
  const int i = threadIdx.x;
  __shared__ float bv[16];
  if (i < 16) { bv[i] = 0.f; Bvec[(size_t)C * 16 + i] = 0.f; }
  __syncthreads();
  for (int k = C - 1; k >= 0; --k) {
    float nv = 0.f;
    if (i < 16) {
      const float* M = chunkM + (size_t)k * 256;
      float mx = NEG_INF;
      #pragma unroll
      for (int j = 0; j < 16; ++j) mx = fmaxf(mx, M[i * 16 + j] + bv[j]);
      float s = 0.f;
      #pragma unroll
      for (int j = 0; j < 16; ++j) s += __expf(M[i * 16 + j] + bv[j] - mx);
      nv = mx + __logf(s);
    }
    __syncthreads();
    if (i < 16) { bv[i] = nv; Bvec[(size_t)k * 16 + i] = nv; }
    __syncthreads();
  }
}

// Phase 3 forward: per-step alphas within each chunk
__global__ void k_fill_f(const float* __restrict__ Avec, const float* __restrict__ ltrs,
                         const float* __restrict__ lpis, float* __restrict__ out_alpha, int T)
{
  const int k = blockIdx.x;
  const int lane = threadIdx.x;
  __shared__ float av[16];
  __shared__ float lt[256];
  if (lane < 16) {
    float a = Avec[(size_t)k * 16 + lane];
    av[lane] = a;
    out_alpha[(size_t)k * LC * 16 + lane] = a;
  }
  __syncthreads();
  int tend = (k + 1) * LC - 1; if (tend > T - 1) tend = T - 1;
  for (int t = k * LC + 1; t <= tend; ++t) {
    #pragma unroll
    for (int q = 0; q < 8; ++q)
      lt[lane + q * 32] = ltrs[(size_t)(t - 1) * 256 + lane + q * 32];
    __syncthreads();
    float nv = 0.f;
    if (lane < 16) {
      float mx = NEG_INF;
      #pragma unroll
      for (int i = 0; i < 16; ++i) mx = fmaxf(mx, av[i] + lt[i * 16 + lane]);
      float s = 0.f;
      #pragma unroll
      for (int i = 0; i < 16; ++i) s += __expf(av[i] + lt[i * 16 + lane] - mx);
      nv = mx + __logf(s) + lpis[(size_t)t * 16 + lane];
      out_alpha[(size_t)t * 16 + lane] = nv;
    }
    __syncthreads();
    if (lane < 16) av[lane] = nv;
    __syncthreads();
  }
}

// Phase 3 backward: per-step betas within each chunk
__global__ void k_fill_b(const float* __restrict__ Bvec, const float* __restrict__ ltrs,
                         const float* __restrict__ lpis, float* __restrict__ out_beta, int T)
{
  const int k = blockIdx.x;
  const int lane = threadIdx.x;
  __shared__ float qv[16];
  __shared__ float lt[256];
  float bvv = 0.f;
  if (lane < 16) bvv = Bvec[(size_t)(k + 1) * 16 + lane];
  int ttop = (k + 1) * LC - 1; if (ttop > T - 1) ttop = T - 1;
  for (int s = ttop; s >= k * LC; --s) {
    if (s == T - 1) {
      if (lane < 16) out_beta[(size_t)s * 16 + lane] = bvv;  // zeros
      continue;
    }
    if (lane < 16) qv[lane] = lpis[(size_t)(s + 1) * 16 + lane] + bvv;
    #pragma unroll
    for (int q = 0; q < 8; ++q)
      lt[lane + q * 32] = ltrs[(size_t)s * 256 + lane + q * 32];
    __syncthreads();
    if (lane < 16) {
      float mx = NEG_INF;
      #pragma unroll
      for (int j = 0; j < 16; ++j) mx = fmaxf(mx, lt[lane * 16 + j] + qv[j]);
      float sm = 0.f;
      #pragma unroll
      for (int j = 0; j < 16; ++j) sm += __expf(lt[lane * 16 + j] + qv[j] - mx);
      bvv = mx + __logf(sm);
      out_beta[(size_t)s * 16 + lane] = bvv;
    }
    __syncthreads();
  }
}

// ---------------------------------------------------------------------------
// Launch
// ---------------------------------------------------------------------------
extern "C" void kernel_launch(void* const* d_in, const int* in_sizes, int n_in,
                              void* d_out, int out_size, void* d_ws, size_t ws_size,
                              hipStream_t stream) {
  (void)n_in; (void)out_size; (void)ws_size;
  const float* s_arr     = (const float*)d_in[0];
  const float* a_arr     = (const float*)d_in[1];
  const float* pW1       = (const float*)d_in[2];
  const float* pb1       = (const float*)d_in[3];
  const float* pW2       = (const float*)d_in[4];
  const float* pb2       = (const float*)d_in[5];
  const float* pW3       = (const float*)d_in[6];
  const float* pb3       = (const float*)d_in[7];
  const float* oW1       = (const float*)d_in[8];
  const float* ob1       = (const float*)d_in[9];
  const float* oW2       = (const float*)d_in[10];
  const float* ob2       = (const float*)d_in[11];
  const float* oW3       = (const float*)d_in[12];
  const float* ob3       = (const float*)d_in[13];
  const float* a_log_std = (const float*)d_in[14];

  const int T = in_sizes[0] / 64;
  const int C = (T - 1 + LC - 1) / LC;

  float* out       = (float*)d_out;
  float* out_alpha = out;
  float* out_beta  = out + (size_t)16 * T;
  float* out_ltrs  = out + (size_t)32 * T;
  float* out_lpis  = out_ltrs + (size_t)256 * (T - 1);
  float* out_ent   = out_lpis + (size_t)16 * T;

  __bf16* wpack   = (__bf16*)d_ws;
  float* fws      = (float*)((char*)d_ws + WPACK_BYTES);
  float* ws_logtr0 = fws;                          // 16
  float* chunkM   = fws + 16;                      // C*256
  float* Avec     = chunkM + (size_t)C * 256;      // (C+1)*16
  float* Bvec     = Avec + (size_t)(C + 1) * 16;   // (C+1)*16

  k_pack<<<(16384 + 255) / 256, 256, 0, stream>>>(pW1, wpack + W1P, 64, 256);
  k_pack<<<(65536 + 255) / 256, 256, 0, stream>>>(pW2, wpack + W2P, 256, 256);
  k_pack<<<(32768 + 255) / 256, 256, 0, stream>>>(pW3, wpack + W3P, 256, 128);
  k_pack<<<(16384 + 255) / 256, 256, 0, stream>>>(oW1, wpack + W1O, 64, 256);
  k_pack<<<(65536 + 255) / 256, 256, 0, stream>>>(oW2, wpack + W2O, 256, 256);
  k_pack<<<(69632 + 255) / 256, 256, 0, stream>>>(oW3, wpack + W3O, 256, 272);

  k_mlp<<<T / 64, 128, 0, stream>>>(s_arr, a_arr, pb1, pb2, pb3, ob1, ob2, ob3,
                                    a_log_std, wpack, out_ltrs, out_lpis, out_ent,
                                    ws_logtr0, T);

  k_chunk<<<C, 256, 0, stream>>>(out_ltrs, out_lpis, chunkM, T);
  k_scan_f<<<1, 32, 0, stream>>>(chunkM, ws_logtr0, out_lpis, Avec, C);
  k_scan_b<<<1, 32, 0, stream>>>(chunkM, Bvec, C);
  k_fill_f<<<C, 32, 0, stream>>>(Avec, out_ltrs, out_lpis, out_alpha, T);
  k_fill_b<<<C, 32, 0, stream>>>(Bvec, out_ltrs, out_lpis, out_beta, T);
}
